// MoA_33784212750966
// MI455X (gfx1250) — matrix operand order repaired and checked by reading, not compiled, not run on hardware
//
#include <hip/hip_runtime.h>
#include <stdint.h>

// ---------------------------------------------------------------------------
// MoA block for MI455X (gfx1250): wave32 + v_wmma_f32_16x16x32_bf16 everywhere.
// ---------------------------------------------------------------------------

typedef __bf16 bf16_t;
typedef bf16_t bf16x16 __attribute__((ext_vector_type(16)));
typedef bf16_t bf16x8  __attribute__((ext_vector_type(8)));
typedef float  f32x8   __attribute__((ext_vector_type(8)));
typedef int    i32x4   __attribute__((ext_vector_type(4)));

#define AS1 __attribute__((address_space(1)))
#define AS3 __attribute__((address_space(3)))

#define D_MODEL 256
#define N_EXP 4
#define N_CHUNK 4
#define DC 64
#define N_HEAD 2
#define DH 32
#define T_LEN 1024
#define NTOK 4096            // B*T = 4*1024
#define NINST 128            // B*C*E*H
#define ATT_SCALE 0.17677669529663687f  // 1/sqrt(32)

#if defined(__has_builtin)
#if __has_builtin(__builtin_amdgcn_global_load_async_to_lds_b128)
#define HAVE_ASYNC_LDS 1
#endif
#endif

// ---------------- WMMA helpers (gfx1250 layouts) ----------------------------

__device__ inline f32x8 wmma_bf16(bf16x16 a, bf16x16 b, f32x8 c) {
  // D = A(16x32) * B(32x16) + C, fp32 accumulate
  return __builtin_amdgcn_wmma_f32_16x16x32_bf16(
      false, a, false, b, (short)0, c, false, false);
}

// A 16x32 bf16 fragment from row-major [row][k] memory, leading dim ld (elems).
// Layout: lanes 0-15 M=lane, K=hi*16+0..15 per half-wave (matches B).
// Both 8-element chunks are 16B-aligned contiguous -> two b128 loads.
__device__ inline bf16x16 load_a_frag(const bf16_t* base, int ld) {
  int lane = threadIdx.x & 31;
  int row  = lane & 15;
  int hi   = lane >> 4;
  const bf16_t* p = base + row * ld + hi * 16;
  bf16x8 lo = *(const bf16x8*)(p);
  bf16x8 up = *(const bf16x8*)(p + 8);
  return __builtin_shufflevector(lo, up, 0, 1, 2, 3, 4, 5, 6, 7,
                                 8, 9, 10, 11, 12, 13, 14, 15);
}

// B 32x16 bf16 fragment where memory holds N rows of K contiguous elements
// (i.e. B[k][n] = mem[n*ld + k]).  Column n = lane&15, K = hi*16 + i.
// 32B contiguous per lane -> two b128 loads.
__device__ inline bf16x16 load_b_frag_n(const bf16_t* base, int ld) {
  int lane = threadIdx.x & 31;
  int n  = lane & 15;
  int hi = lane >> 4;
  return *(const bf16x16*)(base + n * ld + hi * 16);
}

__device__ inline float wred_sum32(float v) {
#pragma unroll
  for (int o = 16; o > 0; o >>= 1) v += __shfl_xor(v, o, 32);
  return v;
}

// ---------------- Kernel W: weight fp32 -> bf16 -----------------------------

__global__ void __launch_bounds__(256)
conv_weights_kernel(const float* __restrict__ qw, const float* __restrict__ pw,
                    const float* __restrict__ ow, bf16_t* __restrict__ qb,
                    bf16_t* __restrict__ pb, bf16_t* __restrict__ ob) {
  int i = blockIdx.x * 256 + threadIdx.x;          // grid covers 65536
  if (i < N_EXP * 3 * DC * DC) qb[i] = (bf16_t)qw[i];  // 49152
  if (i < N_EXP * DC * DC)     pb[i] = (bf16_t)pw[i];  // 16384
  if (i < D_MODEL * D_MODEL)   ob[i] = (bf16_t)ow[i];  // 65536
}

// ---------------- Kernel A: LayerNorm + router softmax ----------------------
// One wave per token; lane handles 8 contiguous channels.

__global__ void __launch_bounds__(256)
ln_router_kernel(const float* __restrict__ x, const float* __restrict__ nw,
                 const float* __restrict__ rw, const float* __restrict__ rb,
                 bf16_t* __restrict__ xnb, float* __restrict__ gate) {
  int wave = threadIdx.x >> 5;
  int lane = threadIdx.x & 31;
  int t = blockIdx.x * 8 + wave;
  const float* xp = x + (size_t)t * D_MODEL + lane * 8;
  float v[8], s = 0.f, ss = 0.f;
#pragma unroll
  for (int i = 0; i < 8; i++) { v[i] = xp[i]; s += v[i]; ss += v[i] * v[i]; }
  s  = wred_sum32(s);
  ss = wred_sum32(ss);
  float mean = s * (1.f / D_MODEL);
  float var  = ss * (1.f / D_MODEL) - mean * mean;
  float rinv = rsqrtf(var + 1e-5f);
  float xn[8];
#pragma unroll
  for (int i = 0; i < 8; i++) {
    xn[i] = (v[i] - mean) * rinv * nw[lane * 8 + i];
    xnb[(size_t)t * D_MODEL + lane * 8 + i] = (bf16_t)xn[i];
  }
  float lg[N_EXP];
#pragma unroll
  for (int e = 0; e < N_EXP; e++) {
    float p = 0.f;
#pragma unroll
    for (int i = 0; i < 8; i++) p += xn[i] * rw[e * D_MODEL + lane * 8 + i];
    lg[e] = wred_sum32(p) + rb[e];
  }
  float m = fmaxf(fmaxf(lg[0], lg[1]), fmaxf(lg[2], lg[3]));
  float den = 0.f;
#pragma unroll
  for (int e = 0; e < N_EXP; e++) den += __expf(lg[e] - m);
  float rden = 1.f / den;
  if (lane < N_EXP) gate[t * N_EXP + lane] = __expf(lg[lane] - m) * rden;
}

// ---------------- Kernel B: QKV projection (WMMA GEMM) ----------------------
// Tasks: 256 token-tiles x (C*E)=16 x 12 n-tiles = 49152 wave tasks.
// Q,K stored row-major [inst][t][dh]; V stored TRANSPOSED [inst][dh][t] so the
// attention P*V B-fragment is a contiguous 32B load.

__global__ void __launch_bounds__(256)
qkv_kernel(const bf16_t* __restrict__ xnb, const bf16_t* __restrict__ wb,
           bf16_t* __restrict__ Qg, bf16_t* __restrict__ Kg,
           bf16_t* __restrict__ VgT) {
  int wave = threadIdx.x >> 5, lane = threadIdx.x & 31, hi = lane >> 4;
  int task = blockIdx.x * 8 + wave;
  int tt = task / 192;
  int rem = task % 192;
  int ce = rem / 12, nt = rem % 12;
  int c = ce >> 2, e = ce & 3;
  int gt0 = tt * 16;
  f32x8 acc = {0.f, 0.f, 0.f, 0.f, 0.f, 0.f, 0.f, 0.f};
#pragma unroll
  for (int kk = 0; kk < 2; kk++) {
    bf16x16 a = load_a_frag(xnb + (size_t)gt0 * D_MODEL + c * DC + kk * 32, D_MODEL);
    bf16x16 b = load_b_frag_n(wb + e * (3 * DC * DC) + (nt * 16) * DC + kk * 32, DC);
    acc = wmma_bf16(a, b, acc);
  }
  int n = nt * 16 + (lane & 15);          // 0..191
  int s = n >> 6, w64 = n & 63, h = w64 >> 5, d = w64 & 31;
  float scl = (s == 0) ? ATT_SCALE : 1.0f;
#pragma unroll
  for (int r = 0; r < 8; r++) {
    int gt = gt0 + r + 8 * hi;
    int bb = gt >> 10, tl = gt & 1023;
    int inst = ((bb * N_CHUNK + c) * N_EXP + e) * N_HEAD + h;
    bf16_t val = (bf16_t)(acc[r] * scl);
    if (s == 0) {
      Qg[((size_t)inst * T_LEN + tl) * DH + d] = val;
    } else if (s == 1) {
      Kg[((size_t)inst * T_LEN + tl) * DH + d] = val;
    } else {
      VgT[((size_t)inst * DH + d) * T_LEN + tl] = val;   // transposed
    }
  }
}

// ---------------- Kernel C: flash attention ---------------------------------
// blockIdx = inst*8 + qblock; 8 waves x 16 queries = 128 queries per WG.
// K tile LDS: [64 keys][32 dh]; V tile LDS: [32 dh][64 keys] (transposed).

__global__ void __launch_bounds__(256)
attn_kernel(const bf16_t* __restrict__ Qg, const bf16_t* __restrict__ Kg,
            const bf16_t* __restrict__ VgT, bf16_t* __restrict__ Ybf) {
  __shared__ __align__(16) bf16_t Kt[64 * DH];        // 4KB
  __shared__ __align__(16) bf16_t VtT[DH * 64];       // 4KB
  __shared__ __align__(16) bf16_t Pt[8 * 16 * 64];    // 16KB
  int wave = threadIdx.x >> 5, lane = threadIdx.x & 31;
  int hi = lane >> 4, n = lane & 15;
  int tid = threadIdx.x;
  int inst = blockIdx.x >> 3;
  int qb   = blockIdx.x & 7;
  int q0   = qb * 128 + wave * 16;
  const size_t ibase = (size_t)inst * T_LEN * DH;

  // per-thread staging addresses (128 bits each per iteration)
  int vd   = tid >> 3;             // dh row of V^T
  int voff = (tid & 7) * 8;        // key offset within 64-key tile

  bf16x16 aQ = load_a_frag(Qg + ibase + (size_t)q0 * DH, DH);
  f32x8 o0 = {0.f, 0.f, 0.f, 0.f, 0.f, 0.f, 0.f, 0.f};
  f32x8 o1 = {0.f, 0.f, 0.f, 0.f, 0.f, 0.f, 0.f, 0.f};
  float mrow[8], lrow[8];
#pragma unroll
  for (int r = 0; r < 8; r++) { mrow[r] = -1e30f; lrow[r] = 0.f; }

  for (int kt = 0; kt < T_LEN / 64; kt++) {
    const bf16_t* ksrc = Kg  + ibase + (size_t)kt * 64 * DH + tid * 8;
    const bf16_t* vsrc = VgT + ibase + (size_t)vd * T_LEN + kt * 64 + voff;
    bf16_t* kdst = &Kt[tid * 8];
    bf16_t* vdst = &VtT[vd * 64 + voff];

    __syncthreads();
#if defined(HAVE_ASYNC_LDS)
    // CDNA5 async global->LDS copy, tracked by ASYNCcnt
    __builtin_amdgcn_global_load_async_to_lds_b128(
        (AS1 i32x4*)(bf16_t*)(uintptr_t)(const uintptr_t)ksrc,
        (AS3 i32x4*)kdst, 0, 0);
    __builtin_amdgcn_global_load_async_to_lds_b128(
        (AS1 i32x4*)(bf16_t*)(uintptr_t)(const uintptr_t)vsrc,
        (AS3 i32x4*)vdst, 0, 0);
    asm volatile("s_wait_asynccnt 0x0" ::: "memory");
#else
    *(i32x4*)kdst = *(const i32x4*)ksrc;
    *(i32x4*)vdst = *(const i32x4*)vsrc;
#endif
    __syncthreads();

    if (kt + 1 < T_LEN / 64) {  // prefetch next tiles -> global_prefetch_b8
      __builtin_prefetch((const void*)(ksrc + 64 * DH), 0, 1);
      __builtin_prefetch((const void*)(vsrc + 64), 0, 1);
    }

    f32x8 sj[4];
    const f32x8 zero = {0.f, 0.f, 0.f, 0.f, 0.f, 0.f, 0.f, 0.f};
#pragma unroll
    for (int j = 0; j < 4; j++) {
      bf16x16 bK = load_b_frag_n(&Kt[(j * 16) * DH], DH);  // S = Q*K^T
      sj[j] = wmma_bf16(aQ, bK, zero);
    }
#pragma unroll
    for (int r = 0; r < 8; r++) {
      float tm = fmaxf(fmaxf(sj[0][r], sj[1][r]), fmaxf(sj[2][r], sj[3][r]));
#pragma unroll
      for (int o = 8; o > 0; o >>= 1) tm = fmaxf(tm, __shfl_xor(tm, o, 32));
      float nm = fmaxf(mrow[r], tm);
      float al = __expf(mrow[r] - nm);
      mrow[r] = nm;
      lrow[r] *= al;
      o0[r] *= al;
      o1[r] *= al;
      float rs = 0.f;
#pragma unroll
      for (int j = 0; j < 4; j++) {
        float p = __expf(sj[j][r] - nm);
        rs += p;
        Pt[(wave * 16 + r + 8 * hi) * 64 + j * 16 + n] = (bf16_t)p;
      }
#pragma unroll
      for (int o = 8; o > 0; o >>= 1) rs += __shfl_xor(rs, o, 32);
      lrow[r] += rs;
    }
    // same-wave LDS RAW before A-fragment reload of P (CDNA5 split counter)
    asm volatile("s_wait_dscnt 0x0" ::: "memory");
#pragma unroll
    for (int kk = 0; kk < 2; kk++) {
      bf16x16 aP = load_a_frag(&Pt[wave * 16 * 64 + kk * 32], 64);
      // B[k=key][n=dh] from V^T LDS ([dh][key] => n-major, ld=64)
      bf16x16 b0 = load_b_frag_n(&VtT[0 * 64 + kk * 32], 64);
      bf16x16 b1 = load_b_frag_n(&VtT[16 * 64 + kk * 32], 64);
      o0 = wmma_bf16(aP, b0, o0);
      o1 = wmma_bf16(aP, b1, o1);
    }
  }
  int h = inst & 1, e = (inst >> 1) & 3, c = (inst >> 3) & 3, b = inst >> 5;
#pragma unroll
  for (int r = 0; r < 8; r++) {
    float inv = 1.0f / lrow[r];
    int t = q0 + r + 8 * hi;
    size_t base = ((size_t)(b * T_LEN + t)) * (N_CHUNK * N_EXP * DC) +
                  c * (N_EXP * DC) + e * DC + h * DH;
    Ybf[base + n]      = (bf16_t)(o0[r] * inv);
    Ybf[base + 16 + n] = (bf16_t)(o1[r] * inv);
  }
}

// ---------------- Kernel D: expert projection + gate combine ----------------
// Tasks: 256 token-tiles x C x (DC/16)=4 n-tiles = 4096 wave tasks.

__global__ void __launch_bounds__(256)
proj_gate_kernel(const bf16_t* __restrict__ Ybf, const bf16_t* __restrict__ pw,
                 const float* __restrict__ pb, const float* __restrict__ gate,
                 bf16_t* __restrict__ xcomb) {
  int wave = threadIdx.x >> 5, lane = threadIdx.x & 31;
  int hi = lane >> 4, nn = lane & 15;
  int task = blockIdx.x * 8 + wave;
  int tt = task >> 4;
  int rem = task & 15;
  int c = rem >> 2, nt = rem & 3;
  int gt0 = tt * 16;
  int n = nt * 16 + nn;                       // 0..63
  f32x8 comb = {0.f, 0.f, 0.f, 0.f, 0.f, 0.f, 0.f, 0.f};
#pragma unroll
  for (int e = 0; e < N_EXP; e++) {
    f32x8 acc = {0.f, 0.f, 0.f, 0.f, 0.f, 0.f, 0.f, 0.f};
#pragma unroll
    for (int kk = 0; kk < 2; kk++) {
      bf16x16 a = load_a_frag(
          Ybf + (size_t)gt0 * 1024 + c * (N_EXP * DC) + e * DC + kk * 32, 1024);
      bf16x16 b = load_b_frag_n(pw + e * DC * DC + (nt * 16) * DC + kk * 32, DC);
      acc = wmma_bf16(a, b, acc);
    }
    float bias = pb[e * DC + n];
#pragma unroll
    for (int r = 0; r < 8; r++) {
      int t = gt0 + r + 8 * hi;
      comb[r] += gate[t * N_EXP + e] * (acc[r] + bias);
    }
  }
#pragma unroll
  for (int r = 0; r < 8; r++) {
    int t = gt0 + r + 8 * hi;
    xcomb[(size_t)t * D_MODEL + c * DC + n] = (bf16_t)comb[r];
  }
}

// ---------------- Kernel E: output projection + residual --------------------
// Tasks: 256 token-tiles x 16 n-tiles = 4096 wave tasks.

__global__ void __launch_bounds__(256)
out_proj_kernel(const bf16_t* __restrict__ xcomb, const bf16_t* __restrict__ owb,
                const float* __restrict__ x, float* __restrict__ out) {
  int wave = threadIdx.x >> 5, lane = threadIdx.x & 31;
  int hi = lane >> 4, nn = lane & 15;
  int task = blockIdx.x * 8 + wave;
  int tt = task >> 4, nt = task & 15;
  int gt0 = tt * 16;
  int n = nt * 16 + nn;
  f32x8 acc = {0.f, 0.f, 0.f, 0.f, 0.f, 0.f, 0.f, 0.f};
#pragma unroll
  for (int kk = 0; kk < 8; kk++) {
    bf16x16 a = load_a_frag(xcomb + (size_t)gt0 * D_MODEL + kk * 32, D_MODEL);
    // out = xcomb @ out_w^T : B[k=d][n=o] = out_w[o][d] (row o, k contiguous)
    bf16x16 b = load_b_frag_n(owb + (nt * 16) * D_MODEL + kk * 32, D_MODEL);
    acc = wmma_bf16(a, b, acc);
  }
#pragma unroll
  for (int r = 0; r < 8; r++) {
    int t = gt0 + r + 8 * hi;
    out[(size_t)t * D_MODEL + n] = x[(size_t)t * D_MODEL + n] + acc[r];
  }
}

// ---------------- launch ----------------------------------------------------

extern "C" void kernel_launch(void* const* d_in, const int* in_sizes, int n_in,
                              void* d_out, int out_size, void* d_ws, size_t ws_size,
                              hipStream_t stream) {
  (void)in_sizes; (void)n_in; (void)out_size; (void)ws_size;
  const float* x        = (const float*)d_in[0];
  const float* norm_w   = (const float*)d_in[1];
  const float* router_w = (const float*)d_in[2];
  const float* router_b = (const float*)d_in[3];
  const float* qkv_w    = (const float*)d_in[4];
  const float* proj_w   = (const float*)d_in[5];
  const float* proj_b   = (const float*)d_in[6];
  const float* out_w    = (const float*)d_in[7];
  float* out = (float*)d_out;

  char* ws = (char*)d_ws;
  size_t off = 0;
  auto alloc = [&](size_t bytes) -> void* {
    void* p = ws + off;
    off += (bytes + 255) & ~(size_t)255;
    return p;
  };
  bf16_t* xnb   = (bf16_t*)alloc((size_t)NTOK * D_MODEL * 2);            // 2 MB
  float*  gate  = (float*)alloc((size_t)NTOK * N_EXP * 4);               // 64 KB
  bf16_t* qwb   = (bf16_t*)alloc((size_t)N_EXP * 3 * DC * DC * 2);       // 96 KB
  bf16_t* pwb   = (bf16_t*)alloc((size_t)N_EXP * DC * DC * 2);           // 32 KB
  bf16_t* owb   = (bf16_t*)alloc((size_t)D_MODEL * D_MODEL * 2);         // 128 KB
  bf16_t* Qg    = (bf16_t*)alloc((size_t)NINST * T_LEN * DH * 2);        // 8 MB
  bf16_t* Kg    = (bf16_t*)alloc((size_t)NINST * T_LEN * DH * 2);        // 8 MB
  bf16_t* VgT   = (bf16_t*)alloc((size_t)NINST * T_LEN * DH * 2);        // 8 MB
  bf16_t* Ybf   = (bf16_t*)alloc((size_t)NTOK * 1024 * 2);               // 8 MB
  bf16_t* xcomb = (bf16_t*)alloc((size_t)NTOK * D_MODEL * 2);            // 2 MB

  conv_weights_kernel<<<256, 256, 0, stream>>>(qkv_w, proj_w, out_w, qwb, pwb, owb);
  ln_router_kernel<<<NTOK / 8, 256, 0, stream>>>(x, norm_w, router_w, router_b, xnb, gate);
  qkv_kernel<<<49152 / 8, 256, 0, stream>>>(xnb, qwb, Qg, Kg, VgT);
  attn_kernel<<<NINST * 8, 256, 0, stream>>>(Qg, Kg, VgT, Ybf);
  proj_gate_kernel<<<4096 / 8, 256, 0, stream>>>(Ybf, pwb, proj_b, gate, xcomb);
  out_proj_kernel<<<4096 / 8, 256, 0, stream>>>(xcomb, owb, x, out);
}
